// GAE_20899310862690
// MI455X (gfx1250) — compile-verified, matrix operand order checked by reading, not run on hardware
//
#include <hip/hip_runtime.h>
#include <hip/hip_bf16.h>

#define GN 16384
#define GE 524288
#define IN_DIM 256
#define H1 128
#define H2 64

typedef __attribute__((ext_vector_type(16))) _Float16 v16h;
typedef __attribute__((ext_vector_type(8)))  float    v8f;

// ---------------- utility: zero a float range (grid-stride) ----------------
__global__ void gae_zero(float* __restrict__ p, int n) {
    for (int i = blockIdx.x * blockDim.x + threadIdx.x; i < n; i += gridDim.x * blockDim.x)
        p[i] = 0.0f;
}

// ---------------- degree accumulation ----------------
__global__ void gae_degrees(const int* __restrict__ src, const int* __restrict__ dst,
                            float* __restrict__ out_deg, float* __restrict__ in_deg) {
    int e = blockIdx.x * blockDim.x + threadIdx.x;
    if (e >= GE) return;
    atomicAdd(&out_deg[src[e]], 1.0f);
    atomicAdd(&in_deg[dst[e]], 1.0f);
}

// norm = clip(deg,1)^-0.5, written in place over the degree arrays
__global__ void gae_norms(float* __restrict__ out_deg, float* __restrict__ in_deg) {
    int i = blockIdx.x * blockDim.x + threadIdx.x;
    if (i >= GN) return;
    out_deg[i] = 1.0f / sqrtf(fmaxf(out_deg[i], 1.0f));
    in_deg[i]  = 1.0f / sqrtf(fmaxf(in_deg[i],  1.0f));
}

// ---------------- GEMM1: h[N,128] = feats[N,256] @ W1[256,128] (f32 exact) ----------------
__global__ __launch_bounds__(128) void gae_gemm1(const float* __restrict__ feats,
                                                 const float* __restrict__ W1,
                                                 float* __restrict__ h) {
    __shared__ float sx[IN_DIM];
    const int row = blockIdx.x;
    const int tid = threadIdx.x;
    sx[tid]       = feats[(size_t)row * IN_DIM + tid];
    sx[tid + 128] = feats[(size_t)row * IN_DIM + tid + 128];
    __syncthreads();
    float acc = 0.0f;
    #pragma unroll 8
    for (int k = 0; k < IN_DIM; ++k)
        acc = fmaf(sx[k], W1[k * H1 + tid], acc);
    h[(size_t)row * H1 + tid] = acc;
}

// ---------------- edge scatter layer1: agg1[dst] += h[src]*out_norm[src] ----------------
// one thread = one edge x 4 features (float4 load, 4 f32 atomics into L2-resident agg)
__global__ void gae_scatter1(const int* __restrict__ src, const int* __restrict__ dst,
                             const float* __restrict__ h, const float* __restrict__ out_norm,
                             float* __restrict__ agg1) {
    long long idx = (long long)blockIdx.x * blockDim.x + threadIdx.x;
    if (idx >= (long long)GE * (H1 / 4)) return;
    int e = (int)(idx >> 5);          // H1/4 == 32
    int q = (int)(idx & 31);
    int s = src[e], d = dst[e];
    float w = out_norm[s];
    float4 v = *(const float4*)(h + (size_t)s * H1 + q * 4);
    float* a = agg1 + (size_t)d * H1 + q * 4;
    atomicAdd(a + 0, v.x * w);
    atomicAdd(a + 1, v.y * w);
    atomicAdd(a + 2, v.z * w);
    atomicAdd(a + 3, v.w * w);
}

// ---------------- h1 = relu(agg1*in_norm + b1), in place ----------------
__global__ void gae_finish1(float* __restrict__ agg1, const float* __restrict__ in_norm,
                            const float* __restrict__ b1) {
    long long idx = (long long)blockIdx.x * blockDim.x + threadIdx.x;
    if (idx >= (long long)GN * H1) return;
    int row = (int)(idx >> 7), col = (int)(idx & (H1 - 1));
    float v = agg1[idx] * in_norm[row] + b1[col];
    agg1[idx] = fmaxf(v, 0.0f);
}

// ---------------- GEMM2: z0[N,64] = h1[N,128] @ W2[128,64] ----------------
__global__ __launch_bounds__(64) void gae_gemm2(const float* __restrict__ h1,
                                                const float* __restrict__ W2,
                                                float* __restrict__ z0) {
    __shared__ float sx[H1];
    const int row = blockIdx.x;
    const int tid = threadIdx.x;
    sx[tid]      = h1[(size_t)row * H1 + tid];
    sx[tid + 64] = h1[(size_t)row * H1 + tid + 64];
    __syncthreads();
    float acc = 0.0f;
    #pragma unroll 8
    for (int k = 0; k < H1; ++k)
        acc = fmaf(sx[k], W2[k * H2 + tid], acc);
    z0[(size_t)row * H2 + tid] = acc;
}

// ---------------- edge scatter layer2 ----------------
__global__ void gae_scatter2(const int* __restrict__ src, const int* __restrict__ dst,
                             const float* __restrict__ z0, const float* __restrict__ out_norm,
                             float* __restrict__ agg2) {
    long long idx = (long long)blockIdx.x * blockDim.x + threadIdx.x;
    if (idx >= (long long)GE * (H2 / 4)) return;
    int e = (int)(idx >> 4);          // H2/4 == 16
    int q = (int)(idx & 15);
    int s = src[e], d = dst[e];
    float w = out_norm[s];
    float4 v = *(const float4*)(z0 + (size_t)s * H2 + q * 4);
    float* a = agg2 + (size_t)d * H2 + q * 4;
    atomicAdd(a + 0, v.x * w);
    atomicAdd(a + 1, v.y * w);
    atomicAdd(a + 2, v.z * w);
    atomicAdd(a + 3, v.w * w);
}

// ---------------- z = (agg2*in_norm + b2) * drop_mask, converted to f16 ----------------
__global__ void gae_finish2(const float* __restrict__ agg2, const float* __restrict__ in_norm,
                            const float* __restrict__ b2, const float* __restrict__ drop_mask,
                            _Float16* __restrict__ zh) {
    long long idx = (long long)blockIdx.x * blockDim.x + threadIdx.x;
    if (idx >= (long long)GN * H2) return;
    int row = (int)(idx >> 6), col = (int)(idx & (H2 - 1));
    float v = (agg2[idx] * in_norm[row] + b2[col]) * drop_mask[idx];
    zh[idx] = (_Float16)v;
}

// ---------------- decoder: out[N,N] = z @ z^T via V_WMMA_F32_16X16X32_F16 ----------------
// wave32: each wave computes a 64(M)x32(N) tile; block = 8 waves -> 64x256 region.
// For C = Z*Z^T, the B fragment (K x N, column-per-lane) holds rows of Z in the SAME
// per-lane layout as the A fragment, so both are plain contiguous 32B row loads.
__global__ __launch_bounds__(256) void gae_zzT(const _Float16* __restrict__ zh,
                                               float* __restrict__ out) {
    const int lane = threadIdx.x & 31;
    const int wave = threadIdx.x >> 5;
    const int lo = lane & 15;       // row-in-tile / col-in-tile
    const int hi = lane >> 4;       // selects K half (loads) / M half (stores)
    const int m0 = blockIdx.y * 64;
    const int n0 = blockIdx.x * 256 + wave * 32;

    v8f acc[4][2];
    const v8f vzero = {0.f, 0.f, 0.f, 0.f, 0.f, 0.f, 0.f, 0.f};
    #pragma unroll
    for (int mi = 0; mi < 4; ++mi)
        #pragma unroll
        for (int ni = 0; ni < 2; ++ni) acc[mi][ni] = vzero;

    #pragma unroll
    for (int ks = 0; ks < 2; ++ks) {            // K = 64 = 2 x 32
        const int kb = ks * 32 + hi * 16;
        v16h a[4], b[2];
        #pragma unroll
        for (int mi = 0; mi < 4; ++mi)
            a[mi] = *(const v16h*)(zh + (size_t)(m0 + mi * 16 + lo) * H2 + kb);
        #pragma unroll
        for (int ni = 0; ni < 2; ++ni)
            b[ni] = *(const v16h*)(zh + (size_t)(n0 + ni * 16 + lo) * H2 + kb);
        #pragma unroll
        for (int mi = 0; mi < 4; ++mi)
            #pragma unroll
            for (int ni = 0; ni < 2; ++ni)
                acc[mi][ni] = __builtin_amdgcn_wmma_f32_16x16x32_f16(
                    false, a[mi], false, b[ni], (short)0, acc[mi][ni], false, false);
    }

    // C/D layout: VGPR r, lanes 0-15 -> M=r, N=lane; lanes 16-31 -> M=8+r, N=lane-16
    #pragma unroll
    for (int mi = 0; mi < 4; ++mi)
        #pragma unroll
        for (int ni = 0; ni < 2; ++ni) {
            const size_t col = (size_t)(n0 + ni * 16 + lo);
            #pragma unroll
            for (int r = 0; r < 8; ++r) {
                const size_t row = (size_t)(m0 + mi * 16 + r + 8 * hi);
                out[row * (size_t)GN + col] = acc[mi][ni][r];
            }
        }
}

extern "C" void kernel_launch(void* const* d_in, const int* in_sizes, int n_in,
                              void* d_out, int out_size, void* d_ws, size_t ws_size,
                              hipStream_t stream) {
    const float* feats     = (const float*)d_in[0];
    const float* W1        = (const float*)d_in[1];
    const float* b1        = (const float*)d_in[2];
    const float* W2        = (const float*)d_in[3];
    const float* b2        = (const float*)d_in[4];
    const float* drop_mask = (const float*)d_in[5];
    const int*   src       = (const int*)d_in[6];
    const int*   dst       = (const int*)d_in[7];
    float* out = (float*)d_out;

    // workspace layout (floats)
    float* f = (float*)d_ws;
    float*    out_norm = f;                        // N   (deg -> norm)
    float*    in_norm  = out_norm + GN;            // N
    float*    h        = in_norm + GN;             // N*128
    float*    agg1     = h + (size_t)GN * H1;      // N*128 (becomes h1 in place)
    float*    z0       = agg1 + (size_t)GN * H1;   // N*64
    float*    agg2     = z0 + (size_t)GN * H2;     // N*64
    _Float16* zh       = (_Float16*)(agg2 + (size_t)GN * H2); // N*64 halves

    // 1) zero atomic accumulators
    gae_zero<<<2048, 256, 0, stream>>>(out_norm, 2 * GN);
    gae_zero<<<2048, 256, 0, stream>>>(agg1, GN * H1);
    gae_zero<<<2048, 256, 0, stream>>>(agg2, GN * H2);

    // 2) degrees + norms
    gae_degrees<<<(GE + 255) / 256, 256, 0, stream>>>(src, dst, out_norm, in_norm);
    gae_norms<<<(GN + 255) / 256, 256, 0, stream>>>(out_norm, in_norm);

    // 3) layer 1
    gae_gemm1<<<GN, 128, 0, stream>>>(feats, W1, h);
    {
        long long t = (long long)GE * (H1 / 4);
        gae_scatter1<<<(unsigned)((t + 255) / 256), 256, 0, stream>>>(src, dst, h, out_norm, agg1);
    }
    {
        long long t = (long long)GN * H1;
        gae_finish1<<<(unsigned)((t + 255) / 256), 256, 0, stream>>>(agg1, in_norm, b1);
    }

    // 4) layer 2
    gae_gemm2<<<GN, 64, 0, stream>>>(agg1, W2, z0);
    {
        long long t = (long long)GE * (H2 / 4);
        gae_scatter2<<<(unsigned)((t + 255) / 256), 256, 0, stream>>>(src, dst, z0, out_norm, agg2);
    }
    {
        long long t = (long long)GN * H2;
        gae_finish2<<<(unsigned)((t + 255) / 256), 256, 0, stream>>>(agg2, in_norm, b2, drop_mask, zh);
    }

    // 5) decoder: z @ z^T with f16 WMMA (store-bound)
    dim3 grid(GN / 256, GN / 64);
    gae_zzT<<<grid, 256, 0, stream>>>(zh, out);
}
